// FrameTransformerDecoder_43654047597150
// MI455X (gfx1250) — matrix operand order repaired and checked by reading, not compiled, not run on hardware
//
#include <hip/hip_runtime.h>
#include <math.h>

// ---------------------------------------------------------------------------
// FrameTransformerDecoder forward for MI455X (gfx1250), wave32 + WMMA f16.
// All GEMMs run on v_wmma_f32_16x16x32_f16 (f16 inputs, f32 accumulate).
// Tile: 128x128 per block (8 waves), K-step 32, vectorized f32->f16 staging.
// B is always staged as [n][k] in LDS (NN sources transposed on store), so
// fragment pulls are contiguous ds_load_b128 for both operand layouts.
// All GEMM shapes in this net are multiples of (128,128,32) -> no guards.
// ---------------------------------------------------------------------------

typedef _Float16 v16h __attribute__((ext_vector_type(16)));
typedef float    v8f  __attribute__((ext_vector_type(8)));
typedef _Float16 h4   __attribute__((ext_vector_type(4)));

#define BINS_C   1024
#define CROP_C   512
#define BATCH_C  4
#define TOK_C    2048          // BATCH * CROP
#define EPS_C    1e-5f
#define INVSCALE 0.03125f      // 1/sqrt(1024)

#define LDA_S 72     // halves per A-row in LDS (32 + pad), 144B = 16B-aligned
#define LDB_S 72     // halves per B-row in LDS ([n][k] layout)

__global__ __launch_bounds__(256) void k_gemm(
    const float* __restrict__ A, const float* __restrict__ B,
    const float* __restrict__ bias, const float* __restrict__ addT,
    float* __restrict__ C,
    int M, int N, int K, int lda, int ldb, int ldc, int ldt,
    long sAb, long sAn, long sBb, long sBn, long sCb, long sCn, long sTb, long sTn,
    int bandCount, int bT, int act, float outScale)
{
    __shared__ _Float16 As[128 * LDA_S];
    __shared__ _Float16 Bs[128 * LDB_S];

    const int z  = blockIdx.z;
    const int bo = z / bandCount, bn = z % bandCount;
    A += bo * sAb + bn * sAn;
    B += bo * sBb + bn * sBn;
    C += bo * sCb + bn * sCn;
    if (addT) addT += bo * sTb + bn * sTn;

    const int m0 = blockIdx.y * 128, n0 = blockIdx.x * 128;
    const int t    = threadIdx.x;
    const int wave = t >> 5, lane = t & 31;
    const int wy = (wave >> 1) * 32;   // 0,32,64,96
    const int wx = (wave & 1) * 64;    // 0,64
    const int lmod = lane & 15;
    const int ahk  = (lane >> 4) << 3;   // A fragment K half-base (0 or 8)
    const int bkb  = (lane >> 4) << 4;   // unused; B fragments also use ahk (0 or 8)

    // fixed staging coordinates (4 quads per thread per tile, fully unrolled)
    const int ar = t >> 3;              // A/B-NT row (stride 32 over its)
    const int ac = (t & 7) << 2;        // A/B-NT k-col
    const int nkk = t >> 5;             // B-NN k-row (stride 8 over its)
    const int nnn = (t & 31) << 2;      // B-NN n-col

    v8f acc[2][4] = {};

    for (int kt = 0; kt < K; kt += 32) {
        // ---- stage A: 128 rows x 32 k (float4 along k) -------------------
#pragma unroll
        for (int it = 0; it < 4; it++) {
            const int r = ar + it * 32;
            const float4 f = *(const float4*)(A + (long)(m0 + r) * lda + kt + ac);
            h4 hv = { (_Float16)f.x, (_Float16)f.y, (_Float16)f.z, (_Float16)f.w };
            *(h4*)&As[r * LDA_S + ac] = hv;
        }
        // ---- stage B into [n][k] layout ----------------------------------
        if (bT) {  // B is [N,K] row-major: float4 along k, direct store
#pragma unroll
            for (int it = 0; it < 4; it++) {
                const int r = ar + it * 32;
                const float4 f = *(const float4*)(B + (long)(n0 + r) * ldb + kt + ac);
                h4 hv = { (_Float16)f.x, (_Float16)f.y, (_Float16)f.z, (_Float16)f.w };
                *(h4*)&Bs[r * LDB_S + ac] = hv;
            }
        } else {   // B is [K,N] row-major: float4 along n, transpose on store
#pragma unroll
            for (int it = 0; it < 4; it++) {
                const int kk = nkk + it * 8;
                const float4 f = *(const float4*)(B + (long)(kt + kk) * ldb + n0 + nnn);
                Bs[(nnn + 0) * LDB_S + kk] = (_Float16)f.x;
                Bs[(nnn + 1) * LDB_S + kk] = (_Float16)f.y;
                Bs[(nnn + 2) * LDB_S + kk] = (_Float16)f.z;
                Bs[(nnn + 3) * LDB_S + kk] = (_Float16)f.w;
            }
        }
        if (kt + 32 < K) {
            __builtin_prefetch(A + (long)(m0 + (t >> 1)) * lda + kt + 32, 0, 1);
            __builtin_prefetch(bT ? B + (long)(n0 + (t >> 1)) * ldb + kt + 32
                                  : B + (long)(kt + 32 + (t & 31)) * ldb + n0, 0, 1);
        }
        __syncthreads();

        // ---- fragments + 2x4 WMMA ----------------------------------------
        v16h af[2], bf[4];
#pragma unroll
        for (int i = 0; i < 2; i++) {
            const int base = (wy + i * 16 + lmod) * LDA_S;
#pragma unroll
            for (int e = 0; e < 8; e++) {
                af[i][e]     = As[base + ahk + e];
                af[i][8 + e] = As[base + ahk + 16 + e];
            }
        }
#pragma unroll
        for (int j = 0; j < 4; j++) {
            const int base = (wx + j * 16 + lmod) * LDB_S;
#pragma unroll
            for (int e = 0; e < 8; e++) {
                bf[j][e]     = Bs[base + ahk + e];
                bf[j][8 + e] = Bs[base + ahk + 16 + e];
            }
        }
#pragma unroll
        for (int i = 0; i < 2; i++)
#pragma unroll
            for (int j = 0; j < 4; j++)
                acc[i][j] = __builtin_amdgcn_wmma_f32_16x16x32_f16(
                    false, af[i], false, bf[j], (short)0, acc[i][j], false, false);
        __syncthreads();
    }

    // ---- epilogue: bias / transposed-add / scale / act -------------------
    const int rofs = (lane >> 4) << 3;   // lanes 16-31 hold rows +8
#pragma unroll
    for (int i = 0; i < 2; i++)
#pragma unroll
        for (int j = 0; j < 4; j++)
#pragma unroll
            for (int r = 0; r < 8; r++) {
                const int row = m0 + wy + i * 16 + rofs + r;
                const int col = n0 + wx + j * 16 + lmod;
                float v = acc[i][j][r];
                if (addT) v += addT[(long)col * ldt + row];   // skew-trick qe^T add
                if (bias) v += bias[col];
                v *= outScale;
                if (act == 1)      v = fmaxf(v, 0.f);
                else if (act == 2) v = v / (1.f + __expf(-v));   // SiLU
                C[(long)row * ldc + col] = v;
            }
}

// ---------------------------------------------------------------------------
// Bottleneck: y[b,h,w] = sum_c x[b,c,h,w] * wc[c]; accumulate sum/sumsq for BN.
// ---------------------------------------------------------------------------
__global__ __launch_bounds__(256) void k_bneck(const float* __restrict__ x,
                                               const float* __restrict__ wc,
                                               float* __restrict__ y,
                                               float* __restrict__ statPair)
{
    __shared__ float s1[256], s2[256];
    const long HW = (long)BINS_C * CROP_C;
    const long t = (long)blockIdx.x * 256 + threadIdx.x;
    float v = 0.f;
    if (t < (long)BATCH_C * HW) {
        const long b = t / HW, r = t % HW;
        const float* xp = x + b * 8 * HW + r;
        float s = 0.f;
#pragma unroll
        for (int c = 0; c < 8; c++) s += xp[c * HW] * wc[c];
        y[t] = s;
        v = s;
    }
    s1[threadIdx.x] = v;
    s2[threadIdx.x] = v * v;
    __syncthreads();
    for (int o = 128; o > 0; o >>= 1) {
        if (threadIdx.x < o) {
            s1[threadIdx.x] += s1[threadIdx.x + o];
            s2[threadIdx.x] += s2[threadIdx.x + o];
        }
        __syncthreads();
    }
    if (threadIdx.x == 0) {
        atomicAdd(&statPair[0], s1[0]);
        atomicAdd(&statPair[1], s2[0]);
    }
}

__global__ void k_bnfin(float* stats, const float* gx, const float* bx,
                        const float* gm, const float* bm)
{
    if (threadIdx.x == 0 && blockIdx.x == 0) {
        const float Nn = (float)((long)BATCH_C * BINS_C * CROP_C);
        float mx = stats[0] / Nn, vx = stats[1] / Nn - mx * mx;
        float sx = gx[0] * rsqrtf(vx + EPS_C);
        stats[4] = sx; stats[5] = bx[0] - mx * sx;
        float mm = stats[2] / Nn, vm = stats[3] / Nn - mm * mm;
        float sm = gm[0] * rsqrtf(vm + EPS_C);
        stats[6] = sm; stats[7] = bm[0] - mm * sm;
    }
}

// BN apply + ReLU + transpose [B,BINS,W] -> token-major [B*W, BINS]
__global__ __launch_bounds__(256) void k_bnapply(const float* __restrict__ y,
                                                 float* __restrict__ xs,
                                                 const float* __restrict__ stats, int sidx)
{
    const long HW = (long)BINS_C * CROP_C;
    const long t = (long)blockIdx.x * 256 + threadIdx.x;
    if (t >= (long)BATCH_C * HW) return;
    const float sc = stats[sidx], sh = stats[sidx + 1];
    const long b = t / HW, r = t % HW;
    const long h = r / CROP_C, w = r % CROP_C;
    xs[(b * CROP_C + w) * BINS_C + h] = fmaxf(y[t] * sc + sh, 0.f);
}

// Row softmax (N=512), one block per row
__global__ __launch_bounds__(256) void k_softmax(const float* __restrict__ a,
                                                 float* __restrict__ o, int N)
{
    __shared__ float red[256];
    const long row = blockIdx.x;
    const float* ar = a + row * (long)N;
    float mx = -1e30f;
    for (int j = threadIdx.x; j < N; j += 256) mx = fmaxf(mx, ar[j]);
    red[threadIdx.x] = mx; __syncthreads();
    for (int s = 128; s > 0; s >>= 1) {
        if (threadIdx.x < s) red[threadIdx.x] = fmaxf(red[threadIdx.x], red[threadIdx.x + s]);
        __syncthreads();
    }
    mx = red[0]; __syncthreads();
    float sm = 0.f;
    for (int j = threadIdx.x; j < N; j += 256) sm += __expf(ar[j] - mx);
    red[threadIdx.x] = sm; __syncthreads();
    for (int s = 128; s > 0; s >>= 1) {
        if (threadIdx.x < s) red[threadIdx.x] += red[threadIdx.x + s];
        __syncthreads();
    }
    const float inv = 1.f / red[0];
    for (int j = threadIdx.x; j < N; j += 256)
        o[row * (long)N + j] = __expf(ar[j] - mx) * inv;
}

// LayerNorm over last dim with fused residual adds; in1 padded to width n1.
__global__ __launch_bounds__(256) void k_ln(const float* __restrict__ in0,
                                            const float* __restrict__ in1, int n1,
                                            const float* __restrict__ in2,
                                            const float* __restrict__ g,
                                            const float* __restrict__ b,
                                            float* __restrict__ out, int N)
{
    __shared__ float buf[4096];
    __shared__ float red[256];
    const long row = blockIdx.x;
    float s = 0.f;
    for (int j = threadIdx.x; j < N; j += 256) {
        float v = in0[row * (long)N + j];
        if (in1 && j < n1) v += in1[row * (long)n1 + j];
        if (in2) v += in2[row * (long)N + j];
        buf[j] = v;
        s += v;
    }
    red[threadIdx.x] = s; __syncthreads();
    for (int o = 128; o > 0; o >>= 1) {
        if (threadIdx.x < o) red[threadIdx.x] += red[threadIdx.x + o];
        __syncthreads();
    }
    const float mean = red[0] / (float)N;
    __syncthreads();
    float s2 = 0.f;
    for (int j = threadIdx.x; j < N; j += 256) {
        const float d = buf[j] - mean;
        s2 += d * d;
    }
    red[threadIdx.x] = s2; __syncthreads();
    for (int o = 128; o > 0; o >>= 1) {
        if (threadIdx.x < o) red[threadIdx.x] += red[threadIdx.x + o];
        __syncthreads();
    }
    const float inv = rsqrtf(red[0] / (float)N + EPS_C);
    for (int j = threadIdx.x; j < N; j += 256)
        out[row * (long)N + j] = (buf[j] - mean) * inv * g[j] + b[j];
}

// Grouped depthwise conv over W (token-major layout), gsz input channels/group.
__global__ __launch_bounds__(256) void k_dwconv(const float* __restrict__ in,
                                                const float* __restrict__ wgt,
                                                float* __restrict__ out,
                                                int Cc, int Kk, int pad, int gsz)
{
    const long t = (long)blockIdx.x * 256 + threadIdx.x;
    const long total = (long)TOK_C * Cc;
    if (t >= total) return;
    const int c = (int)(t % Cc);
    const long bw = t / Cc;
    const int w = (int)(bw % CROP_C);
    const long b = bw / CROP_C;
    const int gbase = (c / gsz) * gsz;
    float s = 0.f;
    for (int i = 0; i < gsz; i++) {
        const float* wrow = wgt + ((long)c * gsz + i) * Kk;
        const int cin = gbase + i;
        for (int k = 0; k < Kk; k++) {
            const int ww = w + k - pad;
            if (ww >= 0 && ww < CROP_C)
                s += in[(b * CROP_C + ww) * (long)Cc + cin] * wrow[k];
        }
    }
    out[t] = s;
}

// Final transpose token-major [B*W, BINS] -> out [B,1,BINS,W]
__global__ __launch_bounds__(256) void k_out(const float* __restrict__ xs,
                                             float* __restrict__ out)
{
    const long HW = (long)BINS_C * CROP_C;
    const long t = (long)blockIdx.x * 256 + threadIdx.x;
    if (t >= (long)BATCH_C * HW) return;
    const long b = t / HW, r = t % HW;
    const long h = r / CROP_C, w = r % CROP_C;
    out[t] = xs[(b * CROP_C + w) * BINS_C + h];
}

// ---------------------------------------------------------------------------
extern "C" void kernel_launch(void* const* d_in, const int* in_sizes, int n_in,
                              void* d_out, int out_size, void* d_ws, size_t ws_size,
                              hipStream_t stream)
{
    (void)in_sizes; (void)n_in; (void)out_size; (void)ws_size;
    auto P = [&](int i) { return (const float*)d_in[i]; };
    float* ws  = (float*)d_ws;
    float* out = (float*)d_out;

    const long SZ = 2097152;                 // 2M floats
    float* Abuf = ws;                        // 2M  (y_x -> Q -> dw tmp)
    float* Bbuf = ws + SZ;                   // 2M  (y_m -> K -> dw tmp)
    float* Cbuf = ws + 2 * SZ;               // 2M  xs
    float* Dbuf = ws + 3 * SZ;               // 2M  ms
    float* Ebuf = ws + 4 * SZ;               // 2M  V / hR / h_m1
    float* Fbuf = ws + 5 * SZ;               // 2M  attn-out o / pw out
    float* Gbuf = ws + 6 * SZ;               // 2M  h outputs
    float* Lbuf = ws + 7 * SZ;               // 2M  xs ping-pong
    float* Jbuf = ws + 8 * SZ;               // 8M  hL / dw2 out  (alias: qe)
    float* Kbuf = ws + 12 * SZ;              // 8M  ln2 out / silu out (alias: attn)
    float* Hqe  = Jbuf;                      // [16,512,512]
    float* Iat  = Kbuf;                      // [16,512,512]
    float* stats = ws + 16 * SZ;             // 8 floats

    hipMemsetAsync(stats, 0, 8 * sizeof(float), stream);

    const long tot = (long)BATCH_C * BINS_C * CROP_C;
    const int nb = (int)((tot + 255) / 256);

    auto gemm = [&](const float* Ag, const float* Bg, const float* bias, const float* addT,
                    float* Cg, int M, int N, int Kd, int lda, int ldb, int ldc, int ldt,
                    long sAb, long sAn, long sBb, long sBn, long sCb, long sCn,
                    long sTb, long sTn, int batches, int bandCount, int bT, int act,
                    float oscale) {
        dim3 g((unsigned)(N / 128), (unsigned)(M / 128), (unsigned)batches);
        k_gemm<<<g, 256, 0, stream>>>(Ag, Bg, bias, addT, Cg, M, N, Kd, lda, ldb, ldc, ldt,
                                      sAb, sAn, sBb, sBn, sCb, sCn, sTb, sTn,
                                      bandCount, bT, act, oscale);
    };

    // Multiband attention; writes pre-softmax logits to aOut and projected h to hOut.
    auto attention = [&](int pb, const float* xq, const float* xkv, float* aOut, float* hOut) {
        // Q/K/V projections (NT GEMMs, weights [1024,1024])
        gemm(xq,  P(pb + 0), P(pb + 1), nullptr, Abuf, TOK_C, 1024, 1024, 1024, 1024, 1024, 0,
             0, 0, 0, 0, 0, 0, 0, 0, 1, 1, 1, 0, 1.f);
        gemm(xkv, P(pb + 2), P(pb + 3), nullptr, Bbuf, TOK_C, 1024, 1024, 1024, 1024, 1024, 0,
             0, 0, 0, 0, 0, 0, 0, 0, 1, 1, 1, 0, 1.f);
        gemm(xkv, P(pb + 4), P(pb + 5), nullptr, Ebuf, TOK_C, 1024, 1024, 1024, 1024, 1024, 0,
             0, 0, 0, 0, 0, 0, 0, 0, 1, 1, 1, 0, 1.f);
        // qe[z] = q_band @ er   (er is [256,512] "NN"), z = (b,band), 16 batches
        gemm(Abuf, P(pb + 8), nullptr, nullptr, Hqe, 512, 512, 256, 1024, 512, 512, 0,
             (long)512 * 1024, 256, 0, 0, (long)1048576, 262144, 0, 0,
             16, 4, 0, 0, 1.f);
        // a[z] = (q_band @ k_band^T + qe[z]^T) / sqrt(h)  -> d_out region
        gemm(Abuf, Bbuf, nullptr, Hqe, aOut, 512, 512, 256, 1024, 1024, 512, 512,
             (long)512 * 1024, 256, (long)512 * 1024, 256,
             (long)1048576, 262144, (long)1048576, 262144,
             16, 4, 1, 0, INVSCALE);
        // softmax rows
        k_softmax<<<dim3(16 * 512), 256, 0, stream>>>(aOut, Iat, 512);
        // o[z] = attn @ v_band ("NN")  -> token-major slice of Fbuf
        gemm(Iat, Ebuf, nullptr, nullptr, Fbuf, 512, 256, 512, 512, 1024, 1024, 0,
             (long)1048576, 262144, (long)512 * 1024, 256,
             (long)512 * 1024, 256, 0, 0,
             16, 4, 0, 0, 1.f);
        // h = o @ wo^T + bo
        gemm(Fbuf, P(pb + 6), P(pb + 7), nullptr, hOut, TOK_C, 1024, 1024, 1024, 1024, 1024, 0,
             0, 0, 0, 0, 0, 0, 0, 0, 1, 1, 1, 0, 1.f);
    };

    // ---- bottleneck + BN + ReLU + transpose ------------------------------
    k_bneck<<<nb, 256, 0, stream>>>((const float*)d_in[0], P(2), Abuf, stats + 0);
    k_bneck<<<nb, 256, 0, stream>>>((const float*)d_in[1], P(5), Bbuf, stats + 2);
    k_bnfin<<<1, 32, 0, stream>>>(stats, P(3), P(4), P(6), P(7));
    k_bnapply<<<nb, 256, 0, stream>>>(Abuf, Cbuf, stats, 4);   // xs (token-major)
    k_bnapply<<<nb, 256, 0, stream>>>(Bbuf, Dbuf, stats, 6);   // ms

    float* a1 = out + 2097152;
    float* a2 = a1 + 4194304;
    float* a3 = a2 + 4194304;
    float* a4 = a3 + 4194304;

    // ---- attention block 1: self + cross, then LN1 -----------------------
    attention(8,  Cbuf, Cbuf, a1, Gbuf);   // sa1 -> Gbuf
    attention(17, Cbuf, Dbuf, a2, Ebuf);   // ea1 -> Ebuf (V no longer needed)
    k_ln<<<TOK_C, 256, 0, stream>>>(Cbuf, Gbuf, 1024, Ebuf, P(52), P(53), Lbuf, 1024);

    // ---- conv mixer ------------------------------------------------------
    k_dwconv<<<(int)((tot + 255) / 256), 256, 0, stream>>>(Lbuf, P(44), Abuf, 1024, 11, 5, 1);
    gemm(Abuf, P(45), nullptr, nullptr, Jbuf, TOK_C, 4096, 1024, 1024, 1024, 4096, 0,
         0, 0, 0, 0, 0, 0, 0, 0, 1, 1, 1, 1, 1.f);               // hL = relu(pw)
    k_dwconv<<<(int)((tot + 255) / 256), 256, 0, stream>>>(Lbuf, P(46), Bbuf, 1024, 7, 3, 1);
    gemm(Bbuf, P(47), nullptr, nullptr, Ebuf, TOK_C, 1024, 1024, 1024, 1024, 1024, 0,
         0, 0, 0, 0, 0, 0, 0, 0, 1, 1, 1, 0, 1.f);               // hR
    k_ln<<<TOK_C, 256, 0, stream>>>(Jbuf, Ebuf, 1024, nullptr, P(54), P(55), Kbuf, 4096);
    k_dwconv<<<(int)((4 * tot + 255) / 256), 256, 0, stream>>>(Kbuf, P(48), Jbuf, 4096, 7, 3, 2);
    gemm(Jbuf, P(49), nullptr, nullptr, Fbuf, TOK_C, 1024, 4096, 4096, 4096, 1024, 0,
         0, 0, 0, 0, 0, 0, 0, 0, 1, 1, 1, 0, 1.f);               // c2 pointwise
    k_ln<<<TOK_C, 256, 0, stream>>>(Lbuf, Fbuf, 1024, nullptr, P(56), P(57), Cbuf, 1024);

    // ---- attention block 2 (self) + LN4 ----------------------------------
    attention(26, Cbuf, Cbuf, a3, Gbuf);
    k_ln<<<TOK_C, 256, 0, stream>>>(Cbuf, Gbuf, 1024, nullptr, P(58), P(59), Lbuf, 1024);

    // ---- attention block 2 (cross) + LN5 ---------------------------------
    attention(35, Lbuf, Dbuf, a4, Gbuf);
    k_ln<<<TOK_C, 256, 0, stream>>>(Lbuf, Gbuf, 1024, nullptr, P(60), P(61), Cbuf, 1024);

    // ---- FFN: silu(xs @ c3^T) @ c4^T, LN6 --------------------------------
    gemm(Cbuf, P(50), nullptr, nullptr, Kbuf, TOK_C, 4096, 1024, 1024, 1024, 4096, 0,
         0, 0, 0, 0, 0, 0, 0, 0, 1, 1, 1, 2, 1.f);               // silu
    gemm(Kbuf, P(51), nullptr, nullptr, Fbuf, TOK_C, 1024, 4096, 4096, 4096, 1024, 0,
         0, 0, 0, 0, 0, 0, 0, 0, 1, 1, 1, 0, 1.f);
    k_ln<<<TOK_C, 256, 0, stream>>>(Cbuf, Fbuf, 1024, nullptr, P(62), P(63), Lbuf, 1024);

    // ---- final transpose into d_out --------------------------------------
    k_out<<<nb, 256, 0, stream>>>(Lbuf, out);
}
